// EGNN_Spherical_67577015435590
// MI455X (gfx1250) — matrix operand order repaired and checked by reading, not compiled, not run on hardware
//
#include <hip/hip_runtime.h>

// ---------------- problem constants (from reference) ----------------
#define N_NODES 8192
#define N_EDGES 32768
#define MUL_S   32
#define MUL_V   16
#define DIM     80          // 32 + 3*16
#define NUM_RBF 16
#define HID     64
#define WNUM    2560
#define O1      1024        // MUL_S*MUL_S
#define O2      1536        // +MUL_S*MUL_V
#define O3      1792        // +MUL_V*MUL_V
#define O4      2304        // +MUL_V*MUL_S
#define CUTOFF  5.0f
#define ALPHA_C 1.0f        // 5/CUTOFF
#define PI_OVER_CUT 0.6283185307179586f

#define RS32 0.17677669529663687f   // 1/sqrt(32)
#define RS16 0.25f                  // 1/sqrt(16)
#define RS2  0.7071067811865476f    // 1/sqrt(2)
#define RS3  0.5773502691896258f    // 1/sqrt(3)
#define SQ3  1.7320508075688772f

// ---------------- tiling ----------------
#define TILE_E    16                 // edges per workgroup (WMMA M)
#define NT_COL    160                // 2560 / 16 column tiles
#define EW_STRIDE 2568               // ushort stride: 2560 + 8 pad (bank decorrelation)
// LDS: bf16 ew stage (82176B) + f32 geometry arrays (9408B) + node ids (128B)
#define SMEM_BYTES (TILE_E*EW_STRIDE*2 + (48+512+768+256+768)*4 + 32*4)

typedef __attribute__((ext_vector_type(16))) __bf16 v16bf;
typedef __attribute__((ext_vector_type(8)))  float  v8f;

__device__ __forceinline__ unsigned short f2bf_rne(float x) {
  unsigned u = __float_as_uint(x);
  u += 0x7fffu + ((u >> 16) & 1u);            // round-to-nearest-even (cold paths)
  return (unsigned short)(u >> 16);
}
__device__ __forceinline__ unsigned short f2bf_trunc(float x) {
  return (unsigned short)(__float_as_uint(x) >> 16);  // -> ds_store_b16_d16_hi
}
__device__ __forceinline__ float bf2f(unsigned short v) {
  return __uint_as_float(((unsigned)v) << 16);
}

__device__ __forceinline__ v16bf load_bf16x16(const unsigned short* p) {
  union { uint4 q[2]; v16bf v; } u;
  u.q[0] = *reinterpret_cast<const uint4*>(p);
  u.q[1] = *reinterpret_cast<const uint4*>(p + 8);
  return u.v;
}

// ---------------------------------------------------------------------------
// Kernel 1: pack W2 (64x2560 f32, row-major) into bf16 WMMA B-operand layout.
// Storage: [tile t 0..159][khalf kk 0..1][lane 0..31][elem j 0..15] bf16,
// element j of lane L: VGPR v=j/2, pair p=j%2 -> K = kk*32 + 2v + 16*(L>>4) + p,
//                                                N = t*16 + (L&15).
// ---------------------------------------------------------------------------
__global__ void pack_w2_kernel(const float* __restrict__ W2,
                               unsigned short* __restrict__ w2p) {
  int idx = blockIdx.x * blockDim.x + threadIdx.x;
  if (idx >= NT_COL * 2 * 32 * 16) return;
  int j    = idx & 15;
  int lane = (idx >> 4) & 31;
  int kk   = (idx >> 9) & 1;
  int t    = idx >> 10;
  int K = kk * 32 + 2 * (j >> 1) + 16 * (lane >> 4) + (j & 1);
  int Nc = t * 16 + (lane & 15);
  w2p[idx] = f2bf_rne(W2[K * WNUM + Nc]);
}

// ---------------------------------------------------------------------------
// Kernel 2: per-edge geometry + RBF + first MLP layer + silu -> bf16 act[E][64]
// ---------------------------------------------------------------------------
__global__ void rbf_act_kernel(const float* __restrict__ pos,
                               const int* __restrict__ eidx,
                               const float* __restrict__ means,
                               const float* __restrict__ betas,
                               const float* __restrict__ W1,
                               const float* __restrict__ b1,
                               unsigned short* __restrict__ act) {
  int e = blockIdx.x * blockDim.x + threadIdx.x;
  if (e >= N_EDGES) return;
  int s = eidx[e];
  int t = eidx[N_EDGES + e];
  float vx = pos[t * 3 + 0] - pos[s * 3 + 0];
  float vy = pos[t * 3 + 1] - pos[s * 3 + 1];
  float vz = pos[t * 3 + 2] - pos[s * 3 + 2];
  float d = sqrtf(vx * vx + vy * vy + vz * vz + 1e-12f);
  float cut = 0.5f * (__cosf(d * PI_OVER_CUT) + 1.0f) * (d < CUTOFF ? 1.0f : 0.0f);
  float expd = __expf(-ALPHA_C * d);
  float rbf[NUM_RBF];
#pragma unroll
  for (int j = 0; j < NUM_RBF; ++j) {
    float dm = expd - means[j];
    rbf[j] = __expf(-betas[j] * dm * dm) * cut;
  }
  unsigned short* ap = act + e * HID;
#pragma unroll 4
  for (int k = 0; k < HID; ++k) {
    float sum = b1[k];
#pragma unroll
    for (int j = 0; j < NUM_RBF; ++j) sum += rbf[j] * W1[j * HID + k];
    float sil = sum / (1.0f + __expf(-sum));     // silu
    ap[k] = f2bf_rne(sil);
  }
}

// ---------------------------------------------------------------------------
// Kernel 3: zero agg (N*80) and cnt (N) scratch
// ---------------------------------------------------------------------------
__global__ void zero_kernel(float* __restrict__ agg, float* __restrict__ cnt) {
  int idx = blockIdx.x * blockDim.x + threadIdx.x;
  int tot = N_NODES * DIM + N_NODES;
  if (idx >= tot) return;
  if (idx < N_NODES * DIM) agg[idx] = 0.0f;
  else                     cnt[idx - N_NODES * DIM] = 0.0f;
}

// ---------------------------------------------------------------------------
// Kernel 4: fused edge kernel.
//  Per block: 16 edges. 8 waves; each wave owns tiles {wid + 8k}, processed in
//  PAIRS (t, t+8): four v_wmma_f32_16x16x32_bf16 per iteration forming two
//  independent zero-initialized accumulator chains (SRC2=0 literal on the
//  first pair, no bias-splat movs). Bias is folded into the C store, which is
//  a pure truncation -> ds_store_b16_d16_hi (no conversion VALU).
//  ew(16x2560) staged bf16 in LDS (92KB total -> 3 blocks/WGP), then consumed
//  by the five einsums and scattered with global f32 atomics.
// ---------------------------------------------------------------------------
__global__ __launch_bounds__(256) void edge_kernel(
    const float* __restrict__ h, const float* __restrict__ pos,
    const int* __restrict__ eidx, const float* __restrict__ b2,
    const unsigned short* __restrict__ act,
    const unsigned short* __restrict__ w2p,
    float* __restrict__ agg, float* __restrict__ cnt) {
  extern __shared__ char smem[];
  unsigned short* ewl = (unsigned short*)smem;               // 16 x EW_STRIDE bf16
  float* y1v   = (float*)(smem + TILE_E * EW_STRIDE * 2);    // 16 x 3
  float* s_src = y1v + 48;                                   // 16 x 32
  float* v_src = s_src + 512;                                // 16 x 48
  float* dotv  = v_src + 768;                                // 16 x 16
  float* crv   = dotv + 256;                                 // 16 x 48
  int*   srcn  = (int*)(crv + 768);                          // 16
  int*   dstn  = srcn + 16;                                  // 16

  const int tid  = threadIdx.x;
  const int lane = tid & 31;
  const int wid  = tid >> 5;
  const int e0   = blockIdx.x * TILE_E;
  const int row  = lane & 15;     // M row within tile / N col within tile
  const int half = lane >> 4;

  // ---- phase 0a: edge endpoints + unit direction (y1 = sqrt(3)*vec/d) ----
  if (tid < TILE_E) {
    int e = e0 + tid;
    int s = eidx[e];
    int t = eidx[N_EDGES + e];
    srcn[tid] = s; dstn[tid] = t;
    float vx = pos[t * 3 + 0] - pos[s * 3 + 0];
    float vy = pos[t * 3 + 1] - pos[s * 3 + 1];
    float vz = pos[t * 3 + 2] - pos[s * 3 + 2];
    float d = sqrtf(vx * vx + vy * vy + vz * vz + 1e-12f);
    float inv = SQ3 / d;
    y1v[tid * 3 + 0] = vx * inv;
    y1v[tid * 3 + 1] = vy * inv;
    y1v[tid * 3 + 2] = vz * inv;
  }
  __syncthreads();

  // ---- phase 0b: gather source features into LDS ----
  for (int i = tid; i < TILE_E * MUL_S; i += 256) {          // 512
    int e = i >> 5, u = i & 31;
    s_src[i] = h[srcn[e] * DIM + u];
  }
  for (int i = tid; i < TILE_E * 48; i += 256) {             // 768
    int e = i / 48, k = i % 48;
    v_src[i] = h[srcn[e] * DIM + MUL_S + k];
  }

  // ---- GEMM phase: A regs (per-edge activations, contiguous k order) ----
  const unsigned short* arow = act + (e0 + row) * HID;
  v16bf a0 = load_bf16x16(arow + 16 * half);       // K 0..31  (two 16B runs)
  v16bf a1 = load_bf16x16(arow + 32 + 16 * half);  // K 32..63

  const v8f zacc = {0.f, 0.f, 0.f, 0.f, 0.f, 0.f, 0.f, 0.f};

  for (int t = wid; t < NT_COL; t += 16) {         // pairs (t, t+8): 10 iters
    int t2 = t + 8;
    int ncol0 = t  * 16 + row;
    int ncol1 = t2 * 16 + row;
    const unsigned short* bp0 = w2p + (size_t)t  * 1024 + lane * 16;
    const unsigned short* bp1 = w2p + (size_t)t2 * 1024 + lane * 16;
    // keep next iteration's B tiles flowing into WGP$ (speculative, droppable)
    __builtin_prefetch(bp0 + 16 * 1024, 0, 0);
    __builtin_prefetch(bp1 + 16 * 1024, 0, 0);
    v16bf b00 = load_bf16x16(bp0);          // tile t,  K 0..31
    v16bf b01 = load_bf16x16(bp0 + 512);    // tile t,  K 32..63
    v16bf b10 = load_bf16x16(bp1);          // tile t2, K 0..31
    v16bf b11 = load_bf16x16(bp1 + 512);    // tile t2, K 32..63
    float bias0 = b2[ncol0];
    float bias1 = b2[ncol1];
    // two independent accumulator chains, SRC2=0 literal on the first pair
    v8f c0 = __builtin_amdgcn_wmma_f32_16x16x32_bf16(false, a0, false, b00,
                                                     (short)0, zacc, false, false);
    v8f c1 = __builtin_amdgcn_wmma_f32_16x16x32_bf16(false, a0, false, b10,
                                                     (short)0, zacc, false, false);
    c0 = __builtin_amdgcn_wmma_f32_16x16x32_bf16(false, a1, false, b01,
                                                 (short)0, c0, false, false);
    c1 = __builtin_amdgcn_wmma_f32_16x16x32_bf16(false, a1, false, b11,
                                                 (short)0, c1, false, false);
#pragma unroll
    for (int g = 0; g < 8; ++g) {          // C layout: M = g + 8*half, N = ncol
      ewl[(g + 8 * half) * EW_STRIDE + ncol0] = f2bf_trunc(c0[g] + bias0);
      ewl[(g + 8 * half) * EW_STRIDE + ncol1] = f2bf_trunc(c1[g] + bias1);
    }
  }
  __syncthreads();

  // ---- phase 0c: dot(v, dir) and cross(v, y1)/sqrt(2) ----
  for (int i = tid; i < TILE_E * MUL_V; i += 256) {          // 256 tasks
    int e = i >> 4, u = i & 15;
    const float* v = &v_src[e * 48 + u * 3];
    const float* y = &y1v[e * 3];
    dotv[i] = (v[0] * y[0] + v[1] * y[1] + v[2] * y[2]) * RS3;
    float* cr = &crv[e * 48 + u * 3];
    cr[0] = (v[1] * y[2] - v[2] * y[1]) * RS2;
    cr[1] = (v[2] * y[0] - v[0] * y[2]) * RS2;
    cr[2] = (v[0] * y[1] - v[1] * y[0]) * RS2;
  }
  __syncthreads();

  // ---- phase 2: messages (16 edges x 80 comps) + scatter-add ----
  for (int task = tid; task < TILE_E * DIM; task += 256) {   // 1280 tasks
    int e = task / DIM, c = task % DIM;
    const unsigned short* ew = &ewl[e * EW_STRIDE];
    const float* ss = &s_src[e * MUL_S];
    float val;
    if (c < MUL_S) {
      float p1 = 0.f, p4 = 0.f;
#pragma unroll 8
      for (int u = 0; u < MUL_S; ++u) p1 += ss[u] * bf2f(ew[u * MUL_S + c]);
#pragma unroll 8
      for (int u = 0; u < MUL_V; ++u)
        p4 += dotv[e * MUL_V + u] * bf2f(ew[O3 + u * MUL_S + c]);
      val = (p1 * RS32 + p4 * RS16) * RS2;
    } else {
      int vc = c - MUL_S, w = vc / 3, d = vc % 3;
      float t2 = 0.f, p3 = 0.f, p5 = 0.f;
#pragma unroll 8
      for (int u = 0; u < MUL_S; ++u) t2 += ss[u] * bf2f(ew[O1 + u * MUL_V + w]);
#pragma unroll 8
      for (int u = 0; u < MUL_V; ++u) {
        p3 += v_src[e * 48 + u * 3 + d] * bf2f(ew[O2 + u * MUL_V + w]);
        p5 += crv[e * 48 + u * 3 + d] * bf2f(ew[O4 + u * MUL_V + w]);
      }
      float p2 = t2 * y1v[e * 3 + d];
      val = (p2 * RS32 + (p3 + p5) * RS16) * RS3;
    }
    atomicAdd(&agg[(size_t)dstn[e] * DIM + c], val);
  }
  if (tid < TILE_E) atomicAdd(&cnt[dstn[tid]], 1.0f);
}

// ---------------------------------------------------------------------------
// Kernel 5: out = agg/max(cnt,1) + self-interaction + h
// ---------------------------------------------------------------------------
__global__ void finalize_kernel(const float* __restrict__ h,
                                const float* __restrict__ Wss,
                                const float* __restrict__ Wvv,
                                const float* __restrict__ agg,
                                const float* __restrict__ cnt,
                                float* __restrict__ out) {
  int idx = blockIdx.x * blockDim.x + threadIdx.x;
  if (idx >= N_NODES * DIM) return;
  int n = idx / DIM, c = idx % DIM;
  float a = agg[idx] / fmaxf(cnt[n], 1.0f);
  float self;
  if (c < MUL_S) {
    float s = 0.f;
#pragma unroll 8
    for (int u = 0; u < MUL_S; ++u) s += h[n * DIM + u] * Wss[u * MUL_S + c];
    self = s * RS32;
  } else {
    int vc = c - MUL_S, w = vc / 3, d = vc % 3;
    float s = 0.f;
#pragma unroll 8
    for (int u = 0; u < MUL_V; ++u)
      s += h[n * DIM + MUL_S + u * 3 + d] * Wvv[u * MUL_V + w];
    self = s * RS16;
  }
  out[idx] = a + self + h[idx];
}

// ---------------------------------------------------------------------------
extern "C" void kernel_launch(void* const* d_in, const int* in_sizes, int n_in,
                              void* d_out, int out_size, void* d_ws, size_t ws_size,
                              hipStream_t stream) {
  const float* h     = (const float*)d_in[0];
  const float* pos   = (const float*)d_in[1];
  const int*   eidx  = (const int*)d_in[2];
  const float* means = (const float*)d_in[3];
  const float* betas = (const float*)d_in[4];
  const float* W1    = (const float*)d_in[5];
  const float* b1    = (const float*)d_in[6];
  const float* W2    = (const float*)d_in[7];
  const float* b2    = (const float*)d_in[8];
  const float* Wss   = (const float*)d_in[9];
  const float* Wvv   = (const float*)d_in[10];
  float* out = (float*)d_out;

  // workspace carve (256B aligned)
  char* ws = (char*)d_ws;
  unsigned short* act  = (unsigned short*)ws;                        // E*64 bf16 = 4 MB
  unsigned short* w2p  = (unsigned short*)(ws + 4 * 1024 * 1024);    // 320 KB packed W2
  float*          agg  = (float*)(ws + 4 * 1024 * 1024 + 512 * 1024);  // N*80 f32
  float*          cnt  = agg + (size_t)N_NODES * DIM;                // N f32

  (void)hipFuncSetAttribute((const void*)edge_kernel,
                            hipFuncAttributeMaxDynamicSharedMemorySize,
                            SMEM_BYTES);

  pack_w2_kernel<<<(NT_COL * 1024 + 255) / 256, 256, 0, stream>>>(W2, w2p);
  rbf_act_kernel<<<(N_EDGES + 255) / 256, 256, 0, stream>>>(
      pos, eidx, means, betas, W1, b1, act);
  zero_kernel<<<(N_NODES * DIM + N_NODES + 255) / 256, 256, 0, stream>>>(agg, cnt);
  edge_kernel<<<N_EDGES / TILE_E, 256, SMEM_BYTES, stream>>>(
      h, pos, eidx, b2, act, w2p, agg, cnt);
  finalize_kernel<<<(N_NODES * DIM + 255) / 256, 256, 0, stream>>>(
      h, Wss, Wvv, agg, cnt, out);
}
